// LayerGINEncoder_8272107012813
// MI455X (gfx1250) — matrix operand order repaired and checked by reading, not compiled, not run on hardware
//
#include <hip/hip_runtime.h>

#define D      128
#define AP     136      // LDS row pitch in bf16 elems (breaks 256B bank stride)
#define RPB    128      // rows per block
#define NGRAPH 64
#define LN_EPS 1e-5f

typedef __attribute__((ext_vector_type(16))) __bf16 v16bf;
typedef __attribute__((ext_vector_type(8)))  float  v8f;

union FragBF { v16bf v; unsigned int u[8]; };

__device__ __forceinline__ unsigned short f2bf(float f) {
  // round-to-nearest-even f32 -> bf16
  unsigned int u = __float_as_uint(f);
  u += 0x7FFFu + ((u >> 16) & 1u);
  return (unsigned short)(u >> 16);
}

// -------- fused (h[+agg]) @ W + b, ReLU, optional LayerNorm; in-place safe ----
template<bool ADD_AGG, bool DO_LN>
__global__ __launch_bounds__(256)
void gin_gemm(const float* __restrict__ X, const float* __restrict__ Agg,
              const float* __restrict__ W, const float* __restrict__ bias,
              const float* __restrict__ gamma, const float* __restrict__ beta,
              float* __restrict__ Out, int n_rows)
{
  __shared__ unsigned short Abf[RPB * AP];  // input rows, bf16
  __shared__ unsigned short Bt [D   * AP];  // W transposed, bf16: Bt[n][k]
  const int tid = threadIdx.x;
  const int rb  = blockIdx.x * RPB;

  // stage input rows (zero-pad past n_rows)
  for (int i = tid; i < RPB * D; i += 256) {
    const int r = i >> 7, c = i & (D - 1);
    const int gr = rb + r;
    float v = 0.0f;
    if (gr < n_rows) {
      v = X[(size_t)gr * D + c];
      if (ADD_AGG) v += Agg[(size_t)gr * D + c];
    }
    Abf[r * AP + c] = f2bf(v);
  }
  // stage W transposed so B-fragment K-pairs are contiguous
  for (int i = tid; i < D * D; i += 256) {
    const int k = i >> 7, n = i & (D - 1);
    Bt[n * AP + k] = f2bf(W[i]);
  }
  __syncthreads();

  const int lane = tid & 31;
  const int wave = tid >> 5;
  const int g16  = lane >> 4;     // lane half: 0 or 1
  const int ln   = lane & 15;
  const int arow = wave * 16 + ln;

  const v8f vzero = {0.f,0.f,0.f,0.f,0.f,0.f,0.f,0.f};
  v8f acc[8];
  #pragma unroll
  for (int nt = 0; nt < 8; ++nt) acc[nt] = vzero;

  #pragma unroll
  for (int kt = 0; kt < 4; ++kt) {
    // A fragment 16x32 bf16 (ISA layout: VGPR v holds K-pair per lane half)
    FragBF a;
    #pragma unroll
    for (int v = 0; v < 8; ++v) {
      const int k0 = kt * 32 + ((v >> 2) << 4) + ((v & 3) << 1) + (g16 << 3);
      a.u[v] = *(const unsigned int*)&Abf[arow * AP + k0];
    }
    #pragma unroll
    for (int nt = 0; nt < 8; ++nt) {
      // B fragment 32x16 bf16: lane = column, K-pairs contiguous in Bt
      FragBF b;
      #pragma unroll
      for (int v = 0; v < 8; ++v) {
        const int k0 = kt * 32 + ((v >> 2) << 4) + ((v & 3) << 1) + (g16 << 3);
        b.u[v] = *(const unsigned int*)&Bt[(nt * 16 + ln) * AP + k0];
      }
      acc[nt] = __builtin_amdgcn_wmma_f32_16x16x32_bf16(
          false, a.v, false, b.v, (short)0, acc[nt], false, false);
    }
  }

  // epilogue: bias + ReLU in fp32 accumulators
  float bia[8], gam[8], bet[8];
  #pragma unroll
  for (int nt = 0; nt < 8; ++nt) {
    const int n = nt * 16 + ln;
    bia[nt] = bias[n];
    if (DO_LN) { gam[nt] = gamma[n]; bet[nt] = beta[n]; }
  }
  #pragma unroll
  for (int nt = 0; nt < 8; ++nt) {
    #pragma unroll
    for (int v = 0; v < 8; ++v)
      acc[nt][v] = fmaxf(acc[nt][v] + bia[nt], 0.0f);
  }

  // layernorm: row stats across 8 tiles + 16-lane column group (in-register)
  float mu[8], rs[8];
  if (DO_LN) {
    #pragma unroll
    for (int v = 0; v < 8; ++v) {
      float s = 0.0f, q = 0.0f;
      #pragma unroll
      for (int nt = 0; nt < 8; ++nt) { const float t = acc[nt][v]; s += t; q += t * t; }
      #pragma unroll
      for (int off = 1; off < 16; off <<= 1) {
        s += __shfl_xor(s, off, 32);
        q += __shfl_xor(q, off, 32);
      }
      const float m = s * (1.0f / D);
      mu[v] = m;
      rs[v] = rsqrtf(q * (1.0f / D) - m * m + LN_EPS);
    }
  }

  // store (C/D layout: row = v + 8*g16, col = nt*16 + ln)
  #pragma unroll
  for (int v = 0; v < 8; ++v) {
    const int gr = rb + wave * 16 + (g16 << 3) + v;
    if (gr < n_rows) {
      #pragma unroll
      for (int nt = 0; nt < 8; ++nt) {
        float o = acc[nt][v];
        if (DO_LN) o = (o - mu[v]) * rs[v] * gam[nt] + bet[nt];
        Out[(size_t)gr * D + nt * 16 + ln] = o;
      }
    }
  }
}

// -------- edge scatter-add: one wave per edge, float4 per lane ---------------
__global__ __launch_bounds__(256)
void gin_scatter(const float* __restrict__ h, const int* __restrict__ src,
                 const int* __restrict__ dst, float* __restrict__ agg, int n_edges)
{
  const int e = blockIdx.x * 8 + (threadIdx.x >> 5);
  if (e >= n_edges) return;
  const int lane = threadIdx.x & 31;
  const int s = src[e], d = dst[e];
  const float4 val = *(const float4*)&h[(size_t)s * D + lane * 4];
  float* o = &agg[(size_t)d * D + lane * 4];
  atomicAdd(o + 0, val.x);
  atomicAdd(o + 1, val.y);
  atomicAdd(o + 2, val.z);
  atomicAdd(o + 3, val.w);
}

__global__ void zero_f4(float4* __restrict__ p, int n4) {
  const int i = blockIdx.x * 256 + threadIdx.x;
  if (i < n4) p[i] = make_float4(0.f, 0.f, 0.f, 0.f);
}

__global__ void pool_count(const int* __restrict__ batch, int* __restrict__ counts, int n) {
  const int i = blockIdx.x * 256 + threadIdx.x;
  if (i < n) atomicAdd(&counts[batch[i]], 1);
}

__global__ void pool_sum(const float* __restrict__ h, const int* __restrict__ batch,
                         float* __restrict__ sums, int n) {
  const int i = blockIdx.x * 256 + threadIdx.x;
  const int node = i >> 7;
  if (node < n) {
    const int d = i & (D - 1);
    atomicAdd(&sums[(size_t)batch[node] * D + d], h[i]);
  }
}

__global__ void pool_final(const float* __restrict__ sums, const int* __restrict__ counts,
                           float* __restrict__ out) {
  const int i = blockIdx.x * 256 + threadIdx.x;
  if (i < NGRAPH * D) {
    const int g = i >> 7;
    out[i] = sums[i] / fmaxf((float)counts[g], 1.0f);
  }
}

extern "C" void kernel_launch(void* const* d_in, const int* in_sizes, int n_in,
                              void* d_out, int out_size, void* d_ws, size_t ws_size,
                              hipStream_t stream)
{
  const float* x          = (const float*)d_in[0];
  const int*   edge_index = (const int*)  d_in[1];
  const int*   batch_ids  = (const int*)  d_in[2];
  const float* proj_W     = (const float*)d_in[3];
  const float* proj_b     = (const float*)d_in[4];
  const float* mlp_W      = (const float*)d_in[5];
  const float* mlp_b      = (const float*)d_in[6];
  const float* ln_g       = (const float*)d_in[7];
  const float* ln_b       = (const float*)d_in[8];

  const int N = in_sizes[0] / D;   // 100000 nodes
  const int E = in_sizes[1] / 2;   // 600000 edges
  const int* src = edge_index;
  const int* dst = edge_index + E;

  // workspace: h [N*D] | agg [N*D] | sums [64*D] | counts [64]  (~103 MB)
  float* h      = (float*)d_ws;
  float* agg    = h + (size_t)N * D;
  float* sums   = agg + (size_t)N * D;
  int*   counts = (int*)(sums + NGRAPH * D);

  const int nblk = (N + RPB - 1) / RPB;
  const int nd4  = (N * D) / 4;

  // proj_in + ReLU
  gin_gemm<false, false><<<nblk, 256, 0, stream>>>(
      x, nullptr, proj_W, proj_b, nullptr, nullptr, h, N);

  for (int layer = 0; layer < 3; ++layer) {
    zero_f4<<<(nd4 + 255) / 256, 256, 0, stream>>>((float4*)agg, nd4);
    gin_scatter<<<(E + 7) / 8, 256, 0, stream>>>(h, src, dst, agg, E);
    gin_gemm<true, true><<<nblk, 256, 0, stream>>>(
        h, agg, mlp_W + (size_t)layer * D * D, mlp_b + (size_t)layer * D,
        ln_g + (size_t)layer * D, ln_b + (size_t)layer * D, h, N);
  }

  // global mean pool
  const int pz4 = (NGRAPH * D + NGRAPH) / 4;  // zero sums + counts together
  zero_f4<<<(pz4 + 255) / 256, 256, 0, stream>>>((float4*)sums, pz4);
  pool_count<<<(N + 255) / 256, 256, 0, stream>>>(batch_ids, counts, N);
  pool_sum<<<(N * D + 255) / 256, 256, 0, stream>>>(h, batch_ids, sums, N);
  pool_final<<<(NGRAPH * D + 255) / 256, 256, 0, stream>>>(sums, counts, (float*)d_out);
}
